// RotAlignKVTranslator_77034533421697
// MI455X (gfx1250) — matrix operand order repaired and checked by reading, not compiled, not run on hardware
//
#include <hip/hip_runtime.h>
#include <cstdint>
#include <cstddef>

// ---------------------------------------------------------------------------
// RotAlignKVTranslator for MI455X (gfx1250, wave32, WMMA bf16 16x16x32 + TDM)
//
// out = quantize( sigmoid(gate) * (src @ [BD(R_s) W BD(R_t)^T]) +
//                 (1-sigmoid(gate)) * tgt )
//
// Stage 1 (prep_w_kernel):  Wp^T[l][n][k] = (R_t @ W_l_block^T @ R_s^T) as bf16
// Stage 2 (kv_gemm_quant):  (T x D) @ (D x D) bf16 WMMA GEMM; B tile staged
//                           into LDS by the Tensor Data Mover (TENSORcnt),
//                           fused sigmoid-gate + 4-bit codebook quantize.
// ---------------------------------------------------------------------------

typedef __attribute__((ext_vector_type(16))) __bf16 v16bf;
typedef __attribute__((ext_vector_type(2)))  __bf16 bf16x2;
typedef __attribute__((ext_vector_type(8)))  float  v8f;
typedef __attribute__((ext_vector_type(4)))  unsigned u32x4;
typedef __attribute__((ext_vector_type(2)))  unsigned u32x2;
typedef __attribute__((ext_vector_type(4)))  float  f32x4;
typedef __attribute__((ext_vector_type(8)))  int    i32x8;
typedef __attribute__((ext_vector_type(4)))  int    i32x4;

#define LNUM 16
#define HNUM 16
#define TNUM 1024
#define DHN  64
#define DNUM 1024   // H * DH

__device__ __forceinline__ unsigned short f2bf(float f) {
  __bf16 h = (__bf16)f;                       // v_cvt bf16 (RNE)
  return __builtin_bit_cast(unsigned short, h);
}
__device__ __forceinline__ unsigned pack2(float a, float b) {
  bf16x2 v = {(__bf16)a, (__bf16)b};          // v_cvt_pk_bf16_f32
  return __builtin_bit_cast(unsigned, v);
}

union FragBF { u32x4 q[2]; v16bf v; };

// A-fragment (16x32 bf16), ISA layout: lane<16: row=lane, K in {0..7, 16..23};
// lane>=16: row=lane-16, K in {8..15, 24..31}. Two 16B LDS reads.
template<int STRIDE>
__device__ __forceinline__ v16bf load_afrag(const unsigned short* S, int rowBase,
                                            int kBase, int lane) {
  int row = rowBase + (lane & 15);
  int kb  = (lane < 16) ? 0 : 8;
  const u32x4* p = (const u32x4*)(S + row * STRIDE + kBase + kb);
  FragBF f;
  f.q[0] = p[0];
  f.q[1] = p[2];   // +16 bf16 = +32 bytes
  return f.v;
}

// B-fragment (32x16 bf16) from S stored [n][k]: lane<16: col=lane, K in {0..7, 16..23};
// lane>=16: col=lane-16, K in {8..15, 24..31}. Two 16B LDS reads.
template<int STRIDE>
__device__ __forceinline__ v16bf load_bfrag(const unsigned short* S, int colBase,
                                            int kBase, int lane) {
  int col  = colBase + (lane & 15);
  int koff = (lane < 16) ? 0 : 8;
  const u32x4* p = (const u32x4*)(S + col * STRIDE + kBase + koff);
  FragBF f;
  f.q[0] = p[0];
  f.q[1] = p[2];
  return f.v;
}

// ---------------------------------------------------------------------------
// Stage 1: Wp^T[l][64*h2+e'][64*h1+i] = sum_{j,e} R_t[e'][e] W[l][64h1+j][64h2+e] R_s[i][j]
// One 128-thread block (4 waves) per (l, h1, h2); two 64x64x64 bf16 WMMA GEMMs.
// ---------------------------------------------------------------------------
__global__ __launch_bounds__(128) void prep_w_kernel(
    const float* __restrict__ W, const float* __restrict__ Rs,
    const float* __restrict__ Rt, unsigned short* __restrict__ WpT) {
  const int h2 = blockIdx.x, h1 = blockIdx.y, l = blockIdx.z;
  const int tid = threadIdx.x;

  __shared__ __align__(16) unsigned short Rt_s[64 * 72];
  __shared__ __align__(16) unsigned short Rs_s[64 * 72];
  __shared__ __align__(16) unsigned short Wb_s[64 * 72];
  __shared__ __align__(16) unsigned short M1_s[64 * 72];

  for (int i = tid; i < 4096; i += 128) {
    int r = i >> 6, c = i & 63;
    Rt_s[r * 72 + c] = f2bf(Rt[i]);
    Rs_s[r * 72 + c] = f2bf(Rs[i]);
    size_t wi = ((size_t)(l * DNUM + h1 * 64 + r)) * DNUM + h2 * 64 + c;
    Wb_s[r * 72 + c] = f2bf(W[wi]);
  }
  __syncthreads();

  const int wave = tid >> 5, lane = tid & 31;
  const int m0 = wave * 16;
  const int hi = lane >> 4, row16 = lane & 15;

  // GEMM1: M1[e'][j] = sum_e Rt[e'][e] * Wb[j][e]   (B[k=e][n=j] = Wb[j][e])
  for (int fn = 0; fn < 4; ++fn) {
    v8f acc = {0.f, 0.f, 0.f, 0.f, 0.f, 0.f, 0.f, 0.f};
    for (int ks = 0; ks < 64; ks += 32) {
      v16bf a = load_afrag<72>(Rt_s, m0, ks, lane);
      v16bf b = load_bfrag<72>(Wb_s, fn * 16, ks, lane);
      acc = __builtin_amdgcn_wmma_f32_16x16x32_bf16(false, a, false, b,
                                                    (short)0, acc, false, false);
    }
    // C layout: VGPR r -> M = m0 + r (+8 for lanes>=16); N = lane&15
    for (int r = 0; r < 8; ++r)
      M1_s[(m0 + r + hi * 8) * 72 + fn * 16 + row16] = f2bf(acc[r]);
  }
  __syncthreads();

  // GEMM2: Out[e'][i] = sum_j M1[e'][j] * Rs[i][j]   (B[k=j][n=i] = Rs[i][j])
  for (int fn = 0; fn < 4; ++fn) {
    v8f acc = {0.f, 0.f, 0.f, 0.f, 0.f, 0.f, 0.f, 0.f};
    for (int ks = 0; ks < 64; ks += 32) {
      v16bf a = load_afrag<72>(M1_s, m0, ks, lane);
      v16bf b = load_bfrag<72>(Rs_s, fn * 16, ks, lane);
      acc = __builtin_amdgcn_wmma_f32_16x16x32_bf16(false, a, false, b,
                                                    (short)0, acc, false, false);
    }
    for (int r = 0; r < 8; ++r) {
      int grow = h2 * 64 + m0 + r + hi * 8;   // output-col dim (n)
      int gcol = h1 * 64 + fn * 16 + row16;   // input dim (k)
      WpT[((size_t)l * DNUM + grow) * DNUM + gcol] = f2bf(acc[r]);
    }
  }
}

// ---------------------------------------------------------------------------
// Stage 2: per layer, (1024 x 1024) @ (1024 x 1024) with fused gate+quantize.
// Block = 256 threads (8 waves, 2x4), tile 128x128, k-step 64 (one full head).
// B tile staged by TDM (tensor_load_to_lds) with padded LDS rows.
// ---------------------------------------------------------------------------
#define KSTEP 64
#define LSTR  80   // padded LDS row stride in bf16 elems (64 + 16)

__global__ __launch_bounds__(256) void kv_gemm_quant_kernel(
    const float* __restrict__ src, const float* __restrict__ tgt,
    const unsigned short* __restrict__ WpT,
    const float* __restrict__ gate, const float* __restrict__ scale,
    const float* __restrict__ bias, const float* __restrict__ codebook,
    const int* __restrict__ layer_map, float* __restrict__ out) {
  const int tb = blockIdx.x * 128;       // T tile
  const int nb = blockIdx.y * 128;       // D (output) tile
  const int l  = blockIdx.z;
  const int tid = threadIdx.x;
  const int lm  = layer_map[l];

  __shared__ __align__(16) unsigned short As[128 * LSTR];
  __shared__ __align__(16) unsigned short Bs[128 * LSTR];

  const int wave = tid >> 5, lane = tid & 31;
  const int wm = wave >> 2, wn = wave & 3;      // 2 x 4 wave grid
  const int hi = lane >> 4, row16 = lane & 15;

  v8f acc[4][2];
  const v8f vzero = {0.f, 0.f, 0.f, 0.f, 0.f, 0.f, 0.f, 0.f};
  for (int i = 0; i < 4; ++i)
    for (int j = 0; j < 2; ++j) acc[i][j] = vzero;

  // A loads: 16 quads of 4 floats per 64-wide row; 16 rows per pass
  const int aq = tid & 15, am = tid >> 4;

  // TDM descriptor pieces that do not change across k iterations
  const unsigned ldsB = (unsigned)(uintptr_t)&Bs[0];   // LDS byte offset
  // group1: workgroup_mask=0 | data_size=1(2B)<<16 | pad_enable<<20 |
  //         pad_interval=4(32 DW)<<22 | pad_amount=7(8 DW)<<25
  const int g1w0 = (1 << 16) | (1 << 20) | (4 << 22) | (7 << 25);
  // tensor_dim0 = 1024 elems: bits[79:48] -> w1[31:16] (low16), w2[15:0] (hi16)
  // tensor_dim1 in bits[111:80] -> w2[31:16] | w3[15:0]: set 1024 rows
  const int g1w1 = (DNUM & 0xFFFF) << 16;               // tensor_dim0 lo
  const int g1w2 = ((DNUM >> 16) & 0xFFFF) | ((DNUM & 0xFFFF) << 16); // td0 hi | td1 lo
  const int g1w3 = ((DNUM >> 16) & 0xFFFF) | (KSTEP << 16);  // td1 hi | tile_dim0=64
  const int g1w4 = 128;                                  // tile_dim1=128, tile_dim2=0
  const int g1w5 = DNUM;                                 // tensor_dim0_stride lo32 = 1024
  const i32x4 gz4 = {0, 0, 0, 0};
  const i32x8 gz8 = {0, 0, 0, 0, 0, 0, 0, 0};

  for (int k0 = 0; k0 < DNUM; k0 += KSTEP) {
    // ---- B tile via Tensor Data Mover: 128 rows x 64 bf16, padded to LSTR
    if (wave == 0) {
      const unsigned short* wptr = WpT + ((size_t)l * DNUM + nb) * DNUM + k0;
      uint64_t ga = (uint64_t)(uintptr_t)wptr;
      u32x4 g0;
      g0.x = 1u;                                   // count=1, user descriptor
      g0.y = ldsB;                                 // lds_addr
      g0.z = (unsigned)(ga & 0xFFFFFFFFu);         // global_addr[31:0]
      g0.w = (unsigned)((ga >> 32) & 0x1FFFFFFu)   // global_addr[56:32]
             | 0x80000000u;                        // type=2 ("image")
      i32x8 g1 = {g1w0, g1w1, g1w2, g1w3, g1w4, g1w5, 0, 0};
      __builtin_amdgcn_tensor_load_to_lds(g0, g1, gz4, gz4, gz8, 0);
    }

    // ---- A tile: src[lm, 0, h, tb..tb+127, 0..63] -> bf16 LDS [m][k]
    {
      int h = k0 >> 6;   // k-step == DH: one full head, dhb == 0
      const float* sbase = src + (((size_t)lm * HNUM + h) * TNUM + tb) * DHN;
      for (int mi = 0; mi < 8; ++mi) {
        int m = am + 16 * mi;
        f32x4 f = *(const f32x4*)(sbase + (size_t)m * DHN + aq * 4);
        u32x2 pk = {pack2(f.x, f.y), pack2(f.z, f.w)};
        *(u32x2*)(&As[m * LSTR + aq * 4]) = pk;
      }
    }

    if (wave == 0) __builtin_amdgcn_s_wait_tensorcnt((short)0);
    __syncthreads();

    for (int kk = 0; kk < KSTEP; kk += 32) {
      v16bf a[4], b[2];
      for (int fm = 0; fm < 4; ++fm)
        a[fm] = load_afrag<LSTR>(As, wm * 64 + fm * 16, kk, lane);
      for (int fn = 0; fn < 2; ++fn)
        b[fn] = load_bfrag<LSTR>(Bs, wn * 32 + fn * 16, kk, lane);
      for (int fm = 0; fm < 4; ++fm)
        for (int fn = 0; fn < 2; ++fn)
          acc[fm][fn] = __builtin_amdgcn_wmma_f32_16x16x32_bf16(
              false, a[fm], false, b[fn], (short)0, acc[fm][fn], false, false);
    }
    __syncthreads();
  }

  // ----- fused epilogue: gate blend + codebook quantize -----
  float cb[16];
#pragma unroll
  for (int i = 0; i < 16; ++i) cb[i] = codebook[i];
  float g = gate[l];
  float alpha = 1.0f / (1.0f + __expf(-g));
  float beta  = 1.0f - alpha;

  for (int fn = 0; fn < 2; ++fn) {
    int dcol = nb + wn * 32 + fn * 16 + row16;   // column in D
    float s    = scale[l * DNUM + dcol];
    float bv   = bias[l * DNUM + dcol];
    float invs = 1.0f / s;
    int h = dcol >> 6, dh = dcol & 63;
    for (int fm = 0; fm < 4; ++fm) {
      for (int r = 0; r < 8; ++r) {
        int t = tb + wm * 64 + fm * 16 + r + hi * 8;
        size_t idx = (((size_t)l * HNUM + h) * TNUM + t) * DHN + dh;
        float y = acc[fm][fn][r];
        float f = alpha * y + beta * tgt[idx];
        float z = (f - bv) * invs;
        // searchsorted(mids, z, 'left'): q = cb[#(mids < z)]
        float q = cb[0];
#pragma unroll
        for (int i = 1; i < 16; ++i) {
          float mid = 0.5f * (cb[i] + cb[i - 1]);
          q = (z > mid) ? cb[i] : q;
        }
        out[idx] = q * s + bv;
      }
    }
  }
}

// ---------------------------------------------------------------------------
extern "C" void kernel_launch(void* const* d_in, const int* in_sizes, int n_in,
                              void* d_out, int out_size, void* d_ws,
                              size_t ws_size, hipStream_t stream) {
  (void)in_sizes; (void)n_in; (void)out_size; (void)ws_size;

  const float* K_src    = (const float*)d_in[0];
  const float* V_src    = (const float*)d_in[1];
  const float* K_tgt    = (const float*)d_in[2];
  const float* V_tgt    = (const float*)d_in[3];
  const float* R_s      = (const float*)d_in[4];
  const float* R_t      = (const float*)d_in[5];
  const float* W_K      = (const float*)d_in[6];
  const float* W_V      = (const float*)d_in[7];
  const float* gate_K   = (const float*)d_in[8];
  const float* gate_V   = (const float*)d_in[9];
  const float* scale_K  = (const float*)d_in[10];
  const float* bias_K   = (const float*)d_in[11];
  const float* scale_V  = (const float*)d_in[12];
  const float* bias_V   = (const float*)d_in[13];
  const float* codebook = (const float*)d_in[14];
  const int*   layer_map= (const int*)d_in[15];

  // workspace: folded bf16 weights Wp^T, L*D*D*2 = 32 MiB, reused K -> V
  unsigned short* WpT = (unsigned short*)d_ws;

  float* outK = (float*)d_out;
  float* outV = outK + (size_t)LNUM * HNUM * TNUM * DHN;

  dim3 pg(HNUM, HNUM, LNUM), pb(128);
  dim3 gg(TNUM / 128, DNUM / 128, LNUM), gb(256);

  prep_w_kernel<<<pg, pb, 0, stream>>>(W_K, R_s, R_t, WpT);
  kv_gemm_quant_kernel<<<gg, gb, 0, stream>>>(K_src, K_tgt, WpT, gate_K,
                                              scale_K, bias_K, codebook,
                                              layer_map, outK);
  prep_w_kernel<<<pg, pb, 0, stream>>>(W_V, R_s, R_t, WpT);
  kv_gemm_quant_kernel<<<gg, gb, 0, stream>>>(V_src, V_tgt, WpT, gate_V,
                                              scale_V, bias_V, codebook,
                                              layer_map, outV);
}
